// HyperbolicTransformerBlock_75660143886520
// MI455X (gfx1250) — matrix operand order repaired and checked by reading, not compiled, not run on hardware
//
#include <hip/hip_runtime.h>
#include <math.h>
#include <stdint.h>

// -------------------------------------------------------------------------
// Hyperbolic transformer block for MI455X (gfx1250, wave32, WMMA).
// GEMMs: f16 operands, double-buffered LDS tiles filled with
// global_load_async_to_lds_b128 (ASYNCcnt), v_wmma_f32_16x16x32_f16 with
// f32 accumulation. Block tile 128x64, wave tile 32x32 (4 accumulators).
// Hyperbolic nonlinearities / norms stay f32 (all vectors have norm < 1).
// -------------------------------------------------------------------------

#define EPSF 1e-15f
#define MAXNF (1.0f - 1e-5f)

typedef __attribute__((ext_vector_type(16))) _Float16 v16h;
typedef __attribute__((ext_vector_type(8)))  _Float16 v8h;
typedef __attribute__((ext_vector_type(8)))  float    v8f;

__device__ __forceinline__ float warpSum(float v) {
#pragma unroll
  for (int o = 16; o > 0; o >>= 1) v += __shfl_xor(v, o, 32);
  return v;
}
__device__ __forceinline__ float warpMax(float v) {
#pragma unroll
  for (int o = 16; o > 0; o >>= 1) v = fmaxf(v, __shfl_xor(v, o, 32));
  return v;
}
__device__ __forceinline__ float blockSum(float v, float* s) {
  v = warpSum(v);
  int lane = threadIdx.x & 31, w = threadIdx.x >> 5;
  __syncthreads();
  if (lane == 0) s[w] = v;
  __syncthreads();
  float r = (threadIdx.x < (blockDim.x >> 5)) ? s[threadIdx.x] : 0.f;
  if (w == 0) { r = warpSum(r); if (lane == 0) s[0] = r; }
  __syncthreads();
  return s[0];
}
__device__ __forceinline__ float blockMax(float v, float* s) {
  v = warpMax(v);
  int lane = threadIdx.x & 31, w = threadIdx.x >> 5;
  __syncthreads();
  if (lane == 0) s[w] = v;
  __syncthreads();
  float r = (threadIdx.x < (blockDim.x >> 5)) ? s[threadIdx.x] : -3.4e38f;
  if (w == 0) { r = warpMax(r); if (lane == 0) s[0] = r; }
  __syncthreads();
  return s[0];
}
__device__ __forceinline__ float artanhf_(float x) {
  return 0.5f * logf((1.f + x) / (1.f - x));
}

// CDNA5 async copy: 16 bytes global -> LDS, tracked by ASYNCcnt.
__device__ __forceinline__ void async_copy16(uint32_t ldsOff, const void* sbase, int byteOff) {
  asm volatile("global_load_async_to_lds_b128 %0, %1, %2"
               :: "v"(ldsOff), "v"(byteOff), "s"(sbase) : "memory");
}
template <int CNT>
__device__ __forceinline__ void wait_async() {
  asm volatile("s_wait_asynccnt %0" :: "i"(CNT) : "memory");
}

// -------------------------------------------------------------------------
// f32 -> f16 bulk convert (8 elements / thread, vectorized)
// -------------------------------------------------------------------------
__global__ __launch_bounds__(256) void cvt_f16_kernel(const float* __restrict__ src,
                                                      _Float16* __restrict__ dst) {
  long i = ((long)blockIdx.x * 256 + threadIdx.x) * 8;
  float4 p0 = ((const float4*)(src + i))[0];
  float4 p1 = ((const float4*)(src + i))[1];
  v8h h;
  h[0] = (_Float16)p0.x; h[1] = (_Float16)p0.y; h[2] = (_Float16)p0.z; h[3] = (_Float16)p0.w;
  h[4] = (_Float16)p1.x; h[5] = (_Float16)p1.y; h[6] = (_Float16)p1.z; h[7] = (_Float16)p1.w;
  *(v8h*)(dst + i) = h;
}

// -------------------------------------------------------------------------
// NT WMMA GEMM, all-f16 operands: C[N,M] = A[N,K] * B[M,K]^T  (f32 accum).
// Block tile 128(N) x 64(M), 8 waves, wave tile 32x32 = 4 accumulators.
// Double-buffered tiles staged via async LDS DMA (3 b128 per thread/tile).
// N multiple of 128, M of 64, K of 32; lda/ldb in halves, rows 16B-aligned.
// -------------------------------------------------------------------------
__global__ __launch_bounds__(256) void gemm_nt_kernel(
    const _Float16* __restrict__ A, int lda, long aB, long aH,
    const _Float16* __restrict__ B, int ldb, long bB, long bH,
    float* __restrict__ C, int ldc, long cB, long cH,
    int K, int zdiv)
{
  __shared__ _Float16 As[2][128][40];  // row stride 80B (16B multiple)
  __shared__ _Float16 Bs[2][64][40];
  const int z = blockIdx.z;
  A += (long)(z / zdiv) * aB + (long)(z % zdiv) * aH;
  B += (long)(z / zdiv) * bB + (long)(z % zdiv) * bH;
  C += (long)(z / zdiv) * cB + (long)(z % zdiv) * cH;

  const int n0 = blockIdx.y * 128, m0 = blockIdx.x * 64;
  const int tid = threadIdx.x;
  const int lane = tid & 31, wave = tid >> 5;
  const int wrow = (wave >> 1) * 32, wcol = (wave & 1) * 32;
  const int lr = lane & 15;
  const int hiHalf = lane >> 4;
  const int koA = hiHalf * 16;  // A frag: lanes 16-31 take K 16..31
  const int kbB = hiHalf * 16;  // B frag: lanes 16-31 take K 16..31

  // staging: thread copies A rows sr, sr+64 and B row sr (16B each)
  const int sr = tid >> 2;
  const int sseg = (tid & 3) * 8;
  const uint32_t ldsA0[2] = {(uint32_t)(uintptr_t)&As[0][sr][sseg],
                             (uint32_t)(uintptr_t)&As[1][sr][sseg]};
  const uint32_t ldsA1[2] = {(uint32_t)(uintptr_t)&As[0][sr + 64][sseg],
                             (uint32_t)(uintptr_t)&As[1][sr + 64][sseg]};
  const uint32_t ldsBb[2] = {(uint32_t)(uintptr_t)&Bs[0][sr][sseg],
                             (uint32_t)(uintptr_t)&Bs[1][sr][sseg]};
  const long aRow0 = (long)(n0 + sr) * lda + sseg;
  const long aRow1 = (long)(n0 + sr + 64) * lda + sseg;
  const long bRow  = (long)(m0 + sr) * ldb + sseg;

  v8f acc00 = {0.f,0.f,0.f,0.f,0.f,0.f,0.f,0.f};
  v8f acc01 = acc00, acc10 = acc00, acc11 = acc00;

  const int KT = K >> 5;
  // prologue: stage tile 0 into buffer 0
  async_copy16(ldsA0[0], A, (int)(aRow0 * 2));
  async_copy16(ldsA1[0], A, (int)(aRow1 * 2));
  async_copy16(ldsBb[0], B, (int)(bRow * 2));

  for (int kt = 0; kt < KT; ++kt) {
    const int cur = kt & 1;
    __syncthreads();  // everyone done reading buf[cur^1] (prev iteration)
    if (kt + 1 < KT) {
      const long k1 = (long)(kt + 1) * 32;
      async_copy16(ldsA0[cur ^ 1], A, (int)((aRow0 + k1) * 2));
      async_copy16(ldsA1[cur ^ 1], A, (int)((aRow1 + k1) * 2));
      async_copy16(ldsBb[cur ^ 1], B, (int)((bRow + k1) * 2));
      wait_async<3>();  // in-order: current tile's 3 copies complete
    } else {
      wait_async<0>();
    }
    __syncthreads();

    v16h a0, a1, b0, b1;
    const _Float16* ap0 = &As[cur][wrow + lr][0];
    const _Float16* ap1 = &As[cur][wrow + 16 + lr][0];
#pragma unroll
    for (int j = 0; j < 8; ++j) {
      a0[j] = ap0[koA + j]; a0[8 + j] = ap0[koA + 8 + j];
      a1[j] = ap1[koA + j]; a1[8 + j] = ap1[koA + 8 + j];
    }
    const _Float16* bp0 = &Bs[cur][wcol + lr][kbB];
    const _Float16* bp1 = &Bs[cur][wcol + 16 + lr][kbB];
#pragma unroll
    for (int j = 0; j < 16; ++j) { b0[j] = bp0[j]; b1[j] = bp1[j]; }

    acc00 = __builtin_amdgcn_wmma_f32_16x16x32_f16(false, a0, false, b0, (short)0, acc00, false, false);
    acc01 = __builtin_amdgcn_wmma_f32_16x16x32_f16(false, a0, false, b1, (short)0, acc01, false, false);
    acc10 = __builtin_amdgcn_wmma_f32_16x16x32_f16(false, a1, false, b0, (short)0, acc10, false, false);
    acc11 = __builtin_amdgcn_wmma_f32_16x16x32_f16(false, a1, false, b1, (short)0, acc11, false, false);
  }

  const int rb0 = n0 + wrow + (hiHalf << 3);
  const int rb1 = rb0 + 16;
  const int c0 = m0 + wcol + lr;
#pragma unroll
  for (int v = 0; v < 8; ++v) {
    C[(long)(rb0 + v) * ldc + c0]      = acc00[v];
    C[(long)(rb0 + v) * ldc + c0 + 16] = acc01[v];
    C[(long)(rb1 + v) * ldc + c0]      = acc10[v];
    C[(long)(rb1 + v) * ldc + c0 + 16] = acc11[v];
  }
}

// -------------------------------------------------------------------------
// NN WMMA GEMM: C[N,M] = A[N,K](f32) * B[K,M](f16).  Used for attn @ tanV.
// Block tile 128x64. A staged with vectorized convert (2x float4 -> one
// b128 LDS store, 2 per thread/tile); B staged async into Bs2[k][m].
// -------------------------------------------------------------------------
__global__ __launch_bounds__(256) void gemm_nn_kernel(
    const float* __restrict__ A, int lda, long aB, long aH,
    const _Float16* __restrict__ B, int ldb, long bB, long bH,
    float* __restrict__ C, int ldc, long cB, long cH,
    int K, int zdiv)
{
  __shared__ _Float16 As[2][128][40];
  __shared__ _Float16 Bs2[2][32][72];  // [k][m], row stride 144B
  const int z = blockIdx.z;
  A += (long)(z / zdiv) * aB + (long)(z % zdiv) * aH;
  B += (long)(z / zdiv) * bB + (long)(z % zdiv) * bH;
  C += (long)(z / zdiv) * cB + (long)(z % zdiv) * cH;

  const int n0 = blockIdx.y * 128, m0 = blockIdx.x * 64;
  const int tid = threadIdx.x;
  const int lane = tid & 31, wave = tid >> 5;
  const int wrow = (wave >> 1) * 32, wcol = (wave & 1) * 32;
  const int lr = lane & 15;
  const int hiHalf = lane >> 4;
  const int koA = hiHalf * 16;
  const int kbB = hiHalf * 16;

  const int sr = tid >> 2;          // A staging rows sr, sr+64
  const int sseg = (tid & 3) * 8;
  const int bk = tid >> 3;          // B staging k 0..31
  const int bm = (tid & 7) * 8;     // B staging m segment
  const uint32_t ldsBb[2] = {(uint32_t)(uintptr_t)&Bs2[0][bk][bm],
                             (uint32_t)(uintptr_t)&Bs2[1][bk][bm]};
  const long bRow = (long)bk * ldb + (m0 + bm);

  v8f acc00 = {0.f,0.f,0.f,0.f,0.f,0.f,0.f,0.f};
  v8f acc01 = acc00, acc10 = acc00, acc11 = acc00;

  const int KT = K >> 5;

  // A convert-staging of one tile chunk into buffer `buf`
  auto stageA = [&](int buf, int k0) {
#pragma unroll
    for (int g = 0; g < 2; ++g) {
      const int r = sr + g * 64;
      const float* apg = A + (long)(n0 + r) * lda + k0 + sseg;
      float4 p0 = ((const float4*)apg)[0];
      float4 p1 = ((const float4*)apg)[1];
      v8h hh;
      hh[0] = (_Float16)p0.x; hh[1] = (_Float16)p0.y; hh[2] = (_Float16)p0.z; hh[3] = (_Float16)p0.w;
      hh[4] = (_Float16)p1.x; hh[5] = (_Float16)p1.y; hh[6] = (_Float16)p1.z; hh[7] = (_Float16)p1.w;
      *(v8h*)&As[buf][r][sseg] = hh;
    }
  };

  // prologue: stage tile 0
  async_copy16(ldsBb[0], B, (int)(bRow * 2));
  stageA(0, 0);

  for (int kt = 0; kt < KT; ++kt) {
    const int cur = kt & 1;
    __syncthreads();
    if (kt + 1 < KT) {
      const int k1 = (kt + 1) * 32;
      async_copy16(ldsBb[cur ^ 1], B, (int)((bRow + (long)k1 * ldb) * 2));
      stageA(cur ^ 1, k1);
      wait_async<1>();
    } else {
      wait_async<0>();
    }
    __syncthreads();

    v16h a0, a1, b0, b1;
    const _Float16* ap0 = &As[cur][wrow + lr][0];
    const _Float16* ap1 = &As[cur][wrow + 16 + lr][0];
#pragma unroll
    for (int j = 0; j < 8; ++j) {
      a0[j] = ap0[koA + j]; a0[8 + j] = ap0[koA + 8 + j];
      a1[j] = ap1[koA + j]; a1[8 + j] = ap1[koA + 8 + j];
    }
#pragma unroll
    for (int j = 0; j < 16; ++j) {
      b0[j] = Bs2[cur][kbB + j][wcol + lr];
      b1[j] = Bs2[cur][kbB + j][wcol + 16 + lr];
    }
    acc00 = __builtin_amdgcn_wmma_f32_16x16x32_f16(false, a0, false, b0, (short)0, acc00, false, false);
    acc01 = __builtin_amdgcn_wmma_f32_16x16x32_f16(false, a0, false, b1, (short)0, acc01, false, false);
    acc10 = __builtin_amdgcn_wmma_f32_16x16x32_f16(false, a1, false, b0, (short)0, acc10, false, false);
    acc11 = __builtin_amdgcn_wmma_f32_16x16x32_f16(false, a1, false, b1, (short)0, acc11, false, false);
  }

  const int rb0 = n0 + wrow + (hiHalf << 3);
  const int rb1 = rb0 + 16;
  const int c0 = m0 + wcol + lr;
#pragma unroll
  for (int v = 0; v < 8; ++v) {
    C[(long)(rb0 + v) * ldc + c0]      = acc00[v];
    C[(long)(rb0 + v) * ldc + c0 + 16] = acc01[v];
    C[(long)(rb1 + v) * ldc + c0]      = acc10[v];
    C[(long)(rb1 + v) * ldc + c0 + 16] = acc11[v];
  }
}

// -------------------------------------------------------------------------
// Row L2 norm (clipped at EPS)
// -------------------------------------------------------------------------
__global__ __launch_bounds__(256) void rownorm_kernel(const float* __restrict__ X,
                                                      int D, float* __restrict__ out) {
  __shared__ float s[32];
  long row = blockIdx.x;
  const float* x = X + row * D;
  float ss = 0.f;
  for (int i = threadIdx.x; i < D; i += 256) { float v = x[i]; ss += v * v; }
  ss = blockSum(ss, s);
  if (threadIdx.x == 0) out[row] = sqrtf(fmaxf(ss, EPSF));
}

// -------------------------------------------------------------------------
// Mobius-linear epilogue (optionally also emits f16 copy of the output).
// -------------------------------------------------------------------------
template <int PER>
__global__ __launch_bounds__(256) void mobius_post_kernel(
    const float* __restrict__ MX, const float* __restrict__ xn,
    const float* __restrict__ bias, float* __restrict__ OUT,
    _Float16* __restrict__ OUT16, int M)
{
  __shared__ float s[32];
  long row = blockIdx.x;
  const float* mx = MX + row * M;
  float* out = OUT + row * M;
  float rm[PER], rb[PER];
  float ss = 0.f, dot = 0.f, bb = 0.f;
#pragma unroll
  for (int i = 0; i < PER; ++i) {
    int idx = threadIdx.x + i * 256;
    float v = mx[idx], b = bias[idx];
    rm[i] = v; rb[i] = b;
    ss += v * v; dot += v * b; bb += b * b;
  }
  ss = blockSum(ss, s); dot = blockSum(dot, s); bb = blockSum(bb, s);
  float mxn = sqrtf(fmaxf(ss, EPSF));
  float xv = xn[row];
  float alpha = artanhf_(fminf(xv, MAXNF));
  float r = tanhf(mxn / xv * alpha);
  float hs = r / mxn;
  float x2 = hs * hs * ss;
  float xy = hs * dot;
  float y2 = bb;
  float cnum = 1.f + 2.f * xy + y2;
  float cy = 1.f - x2;
  float inv = 1.f / fmaxf(1.f + 2.f * xy + x2 * y2, EPSF);
  float oss = 0.f;
#pragma unroll
  for (int i = 0; i < PER; ++i) {
    float o = (cnum * hs * rm[i] + cy * rb[i]) * inv;
    rm[i] = o; oss += o * o;
  }
  oss = blockSum(oss, s);
  float on = sqrtf(fmaxf(oss, EPSF));
  float psc = (on > MAXNF) ? MAXNF / on : 1.f;
  _Float16* o16 = OUT16 ? OUT16 + row * M : nullptr;
#pragma unroll
  for (int i = 0; i < PER; ++i) {
    int idx = threadIdx.x + i * 256;
    float o = psc * rm[i];
    out[idx] = o;
    if (o16) o16[idx] = (_Float16)o;
  }
}

// Per-(row,head) sum of squares over dk=64 for Q and K ([b,h,s] out layout).
__global__ __launch_bounds__(256) void headsq_kernel(
    const float* __restrict__ Q, const float* __restrict__ K,
    float* __restrict__ q2, float* __restrict__ k2)
{
  int item = blockIdx.x * 8 + (threadIdx.x >> 5);
  int lane = threadIdx.x & 31;
  int row = item >> 4, h = item & 15;
  long base = (long)row * 1024 + h * 64;
  float a0 = Q[base + lane], a1 = Q[base + lane + 32];
  float b0 = K[base + lane], b1 = K[base + lane + 32];
  float sq = warpSum(a0 * a0 + a1 * a1);
  float sk = warpSum(b0 * b0 + b1 * b1);
  if (lane == 0) {
    int b = row >> 10, sidx = row & 1023;
    long o = ((long)(b * 16 + h)) * 1024 + sidx;
    q2[o] = sq; k2[o] = sk;
  }
}

// tanV = logmap0(V) per (row, head) over dk=64, f32 in -> f16 out.
__global__ __launch_bounds__(256) void logmap0_head_kernel(
    const float* __restrict__ V, _Float16* __restrict__ T16)
{
  int item = blockIdx.x * 8 + (threadIdx.x >> 5);
  int lane = threadIdx.x & 31;
  int row = item >> 4, h = item & 15;
  long base = (long)row * 1024 + h * 64;
  float a0 = V[base + lane], a1 = V[base + lane + 32];
  float ss = warpSum(a0 * a0 + a1 * a1);
  float n = sqrtf(fmaxf(ss, EPSF));
  float sc = artanhf_(fminf(n, MAXNF)) / n;
  T16[base + lane]      = (_Float16)(sc * a0);
  T16[base + lane + 32] = (_Float16)(sc * a1);
}

// attended = expmap0(rowscale * gemm_out), f32 in-place + f16 copy.
__global__ __launch_bounds__(256) void expmap0_head_kernel(
    float* __restrict__ A_, const float* __restrict__ rowscale,
    _Float16* __restrict__ O16)
{
  int item = blockIdx.x * 8 + (threadIdx.x >> 5);
  int lane = threadIdx.x & 31;
  int row = item >> 4, h = item & 15;
  int b = row >> 10, sidx = row & 1023;
  float rs = rowscale[((long)(b * 16 + h)) * 1024 + sidx];
  long base = (long)row * 1024 + h * 64;
  float a0 = rs * A_[base + lane], a1 = rs * A_[base + lane + 32];
  float ss = warpSum(a0 * a0 + a1 * a1);
  float n = sqrtf(fmaxf(ss, EPSF));
  float sc = tanhf(n) / n;
  float o0 = sc * a0, o1 = sc * a1;
  A_[base + lane] = o0;      A_[base + lane + 32] = o1;
  O16[base + lane] = (_Float16)o0; O16[base + lane + 32] = (_Float16)o1;
}

// -------------------------------------------------------------------------
// Poincare distance -> softmax, in place on raw QK^T scores in d_out.
// Emits rowscale = 1/(sum(attn)+1e-10) for the Einstein midpoint.
// -------------------------------------------------------------------------
__global__ __launch_bounds__(256) void attn_softmax_kernel(
    float* __restrict__ ATT, const float* __restrict__ q2,
    const float* __restrict__ k2, float* __restrict__ rowscale)
{
  __shared__ float s[32];
  int i = blockIdx.x;
  int bh = blockIdx.y;
  long rowo = (long)bh * 1024 + i;
  float* row = ATT + rowo * 1024;
  const float* k2h = k2 + (long)bh * 1024;
  float qi = q2[rowo];
  float oq = 1.f - qi;
  const float scale = 0.125f;  // 1/sqrt(64)
  float r[4];
  float mx = -3.4e38f;
#pragma unroll
  for (int t = 0; t < 4; ++t) {
    int j = threadIdx.x + t * 256;
    float qk = row[j];
    float kj = k2h[j];
    float sq = fmaxf(qi + kj - 2.f * qk, 0.f);
    float den = fmaxf(oq * (1.f - kj), EPSF);
    float arg = fmaxf(1.f + 2.f * sq / den, 1.f + 1e-7f);
    float dist = logf(arg + sqrtf(arg * arg - 1.f));  // arccosh
    float sc = -dist * scale;
    r[t] = sc;
    mx = fmaxf(mx, sc);
  }
  mx = blockMax(mx, s);
  float se = 0.f;
#pragma unroll
  for (int t = 0; t < 4; ++t) { r[t] = expf(r[t] - mx); se += r[t]; }
  se = blockSum(se, s);
  float invs = 1.f / se;
  float s2 = 0.f;
#pragma unroll
  for (int t = 0; t < 4; ++t) { r[t] *= invs; s2 += r[t]; }
  s2 = blockSum(s2, s);
#pragma unroll
  for (int t = 0; t < 4; ++t) row[threadIdx.x + t * 256] = r[t];
  if (threadIdx.x == 0) rowscale[rowo] = 1.f / (s2 + 1e-10f);
}

// -------------------------------------------------------------------------
// Fused residual + hyperbolic layernorm. Optionally emits f16 copy and
// the L2 norm of the output row (feeds next mobius_linear).
// -------------------------------------------------------------------------
__global__ __launch_bounds__(256) void resid_ln_kernel(
    const float* __restrict__ X, const float* __restrict__ Y,
    const float* __restrict__ gamma, const float* __restrict__ beta,
    const float* __restrict__ hbias, float* __restrict__ OUT,
    _Float16* __restrict__ OUT16, float* __restrict__ outNorm)
{
  __shared__ float s[32];
  long row = blockIdx.x;
  const float* x = X + row * 1024;
  const float* y = Y + row * 1024;
  float* out = OUT + row * 1024;
  float rx[4], ry[4];
  float x2 = 0.f, y2 = 0.f, xy = 0.f;
#pragma unroll
  for (int t = 0; t < 4; ++t) {
    int idx = threadIdx.x + t * 256;
    float a = x[idx], b = y[idx];
    rx[t] = a; ry[t] = b;
    x2 += a * a; y2 += b * b; xy += a * b;
  }
  x2 = blockSum(x2, s); y2 = blockSum(y2, s); xy = blockSum(xy, s);
  float cx = 1.f + 2.f * xy + y2, cy = 1.f - x2;
  float inv = 1.f / fmaxf(1.f + 2.f * xy + x2 * y2, EPSF);
  float zz = 0.f;
#pragma unroll
  for (int t = 0; t < 4; ++t) { float z = (cx * rx[t] + cy * ry[t]) * inv; rx[t] = z; zz += z * z; }
  zz = blockSum(zz, s);
  float nz = sqrtf(fmaxf(zz, EPSF));
  float ls = artanhf_(fminf(nz, MAXNF)) / nz;
  float st = 0.f, stt = 0.f;
#pragma unroll
  for (int t = 0; t < 4; ++t) { float v = ls * rx[t]; rx[t] = v; st += v; stt += v * v; }
  st = blockSum(st, s); stt = blockSum(stt, s);
  const float Df = 1024.f;
  float mean = st / Df;
  float var = fmaxf((stt - Df * mean * mean) / (Df - 1.f), 0.f);  // ddof=1
  float istd = 1.f / (sqrtf(var) + 1e-6f);
  float nn = 0.f;
#pragma unroll
  for (int t = 0; t < 4; ++t) {
    int idx = threadIdx.x + t * 256;
    float v = gamma[idx] * (rx[t] - mean) * istd + beta[idx];
    rx[t] = v; nn += v * v;
  }
  nn = blockSum(nn, s);
  float n = sqrtf(fmaxf(nn, EPSF));
  float es = tanhf(n) / n;   // expmap0
  float e2 = es * es * nn;
  float eh = 0.f, hh = 0.f;
#pragma unroll
  for (int t = 0; t < 4; ++t) {
    int idx = threadIdx.x + t * 256;
    float e = es * rx[t]; rx[t] = e;
    float hb = hbias[idx]; ry[t] = hb;
    eh += e * hb; hh += hb * hb;
  }
  eh = blockSum(eh, s); hh = blockSum(hh, s);
  float c1 = 1.f + 2.f * eh + hh, c2 = 1.f - e2;
  float inv2 = 1.f / fmaxf(1.f + 2.f * eh + e2 * hh, EPSF);
  _Float16* o16 = OUT16 ? OUT16 + row * 1024 : nullptr;
  float oss = 0.f;
#pragma unroll
  for (int t = 0; t < 4; ++t) {
    int idx = threadIdx.x + t * 256;
    float o = (c1 * rx[t] + c2 * ry[t]) * inv2;
    out[idx] = o;
    if (o16) o16[idx] = (_Float16)o;
    oss += o * o;
  }
  if (outNorm) {
    oss = blockSum(oss, s);
    if (threadIdx.x == 0) outNorm[row] = sqrtf(fmaxf(oss, EPSF));
  }
}

// HyperbolicActivation on rows of 4096, in place; emits f16 copy + row norm.
__global__ __launch_bounds__(256) void hyp_act_kernel(
    float* __restrict__ Hm, _Float16* __restrict__ H16, float* __restrict__ outNorm)
{
  __shared__ float s[32];
  long row = blockIdx.x;
  float* h = Hm + row * 4096;
  _Float16* h16 = H16 + row * 4096;
  float r[16];
  float ss = 0.f;
#pragma unroll
  for (int t = 0; t < 16; ++t) { float v = h[threadIdx.x + t * 256]; r[t] = v; ss += v * v; }
  ss = blockSum(ss, s);
  float n = sqrtf(fmaxf(ss, EPSF));
  float ls = artanhf_(fminf(n, MAXNF)) / n;
  float rr = 0.f;
#pragma unroll
  for (int t = 0; t < 16; ++t) { float v = fmaxf(ls * r[t], 0.f); r[t] = v; rr += v * v; }
  rr = blockSum(rr, s);
  float n2 = sqrtf(fmaxf(rr, EPSF));
  float es = tanhf(n2) / n2;
#pragma unroll
  for (int t = 0; t < 16; ++t) {
    float o = es * r[t];
    h[threadIdx.x + t * 256] = o;
    h16[threadIdx.x + t * 256] = (_Float16)o;
  }
  if (threadIdx.x == 0) outNorm[row] = sqrtf(fmaxf(es * es * rr, EPSF));
}

// -------------------------------------------------------------------------
extern "C" void kernel_launch(void* const* d_in, const int* in_sizes, int n_in,
                              void* d_out, int out_size, void* d_ws, size_t ws_size,
                              hipStream_t stream)
{
  const int N = 2048, D = 1024, F = 4096, S = 1024, H = 16, BH = 32;

  const float* x   = (const float*)d_in[0];
  const float* wq  = (const float*)d_in[1];  const float* bq  = (const float*)d_in[2];
  const float* wk  = (const float*)d_in[3];  const float* bk  = (const float*)d_in[4];
  const float* wv  = (const float*)d_in[5];  const float* bv  = (const float*)d_in[6];
  const float* wo  = (const float*)d_in[7];  const float* bo  = (const float*)d_in[8];
  const float* w1  = (const float*)d_in[9];  const float* b1  = (const float*)d_in[10];
  const float* w2  = (const float*)d_in[11]; const float* b2  = (const float*)d_in[12];
  const float* g1  = (const float*)d_in[13]; const float* be1 = (const float*)d_in[14];
  const float* hb1 = (const float*)d_in[15];
  const float* g2  = (const float*)d_in[16]; const float* be2 = (const float*)d_in[17];
  const float* hb2 = (const float*)d_in[18];

  float* xout = (float*)d_out;
  float* attn = xout + (size_t)N * D;   // also QK^T scratch (softmax in-place)

  // ---- f32 workspace ----
  float* ws = (float*)d_ws;
  size_t off = 0;
  float* buf0 = ws + off; off += (size_t)N * F;   // GEMM scratch / FFN hidden
  float* bufQ = ws + off; off += (size_t)N * D;   // Q (f32), later x1
  float* bufK = ws + off; off += (size_t)N * D;   // K (f32)
  float* bufV = ws + off; off += (size_t)N * D;   // V (f32)
  float* bufA = ws + off; off += (size_t)N * D;   // attended, later ff
  float* bufO = ws + off; off += (size_t)N * D;   // attn_out
  float* nx = ws + off; off += N;
  float* na = ws + off; off += N;
  float* n1 = ws + off; off += N;
  float* nh = ws + off; off += N;
  float* q2  = ws + off; off += (size_t)BH * S;
  float* k2  = ws + off; off += (size_t)BH * S;
  float* rsc = ws + off; off += (size_t)BH * S;

  // ---- f16 workspace (16B aligned: all counts multiples of 8) ----
  _Float16* hws = (_Float16*)(ws + off);
  size_t hoff = 0;
  _Float16* x16   = hws + hoff; hoff += (size_t)N * D;
  _Float16* q16   = hws + hoff; hoff += (size_t)N * D;
  _Float16* k16   = hws + hoff; hoff += (size_t)N * D;
  _Float16* tv16  = hws + hoff; hoff += (size_t)N * D;   // tanV
  _Float16* at16  = hws + hoff; hoff += (size_t)N * D;   // attended
  _Float16* x116  = hws + hoff; hoff += (size_t)N * D;
  _Float16* h16   = hws + hoff; hoff += (size_t)N * F;
  _Float16* wq16  = hws + hoff; hoff += (size_t)D * D;
  _Float16* wk16  = hws + hoff; hoff += (size_t)D * D;
  _Float16* wv16  = hws + hoff; hoff += (size_t)D * D;
  _Float16* wo16  = hws + hoff; hoff += (size_t)D * D;
  _Float16* w116  = hws + hoff; hoff += (size_t)F * D;
  _Float16* w216  = hws + hoff; hoff += (size_t)D * F;

  dim3 blk(256);
  dim3 gP(D / 64, N / 128, 1);

  // ---- one-time f16 conversions (weights + x) ----
  cvt_f16_kernel<<<(N * D) / 2048, blk, 0, stream>>>(x, x16);
  cvt_f16_kernel<<<(D * D) / 2048, blk, 0, stream>>>(wq, wq16);
  cvt_f16_kernel<<<(D * D) / 2048, blk, 0, stream>>>(wk, wk16);
  cvt_f16_kernel<<<(D * D) / 2048, blk, 0, stream>>>(wv, wv16);
  cvt_f16_kernel<<<(D * D) / 2048, blk, 0, stream>>>(wo, wo16);
  cvt_f16_kernel<<<(F * D) / 2048, blk, 0, stream>>>(w1, w116);
  cvt_f16_kernel<<<(D * F) / 2048, blk, 0, stream>>>(w2, w216);

  // ---- Q/K/V projections ----
  rownorm_kernel<<<N, blk, 0, stream>>>(x, D, nx);
  gemm_nt_kernel<<<gP, blk, 0, stream>>>(x16, D, 0, 0, wq16, D, 0, 0, buf0, D, 0, 0, D, 1);
  mobius_post_kernel<4><<<N, blk, 0, stream>>>(buf0, nx, bq, bufQ, q16, D);
  gemm_nt_kernel<<<gP, blk, 0, stream>>>(x16, D, 0, 0, wk16, D, 0, 0, buf0, D, 0, 0, D, 1);
  mobius_post_kernel<4><<<N, blk, 0, stream>>>(buf0, nx, bk, bufK, k16, D);
  gemm_nt_kernel<<<gP, blk, 0, stream>>>(x16, D, 0, 0, wv16, D, 0, 0, buf0, D, 0, 0, D, 1);
  mobius_post_kernel<4><<<N, blk, 0, stream>>>(buf0, nx, bv, bufV, (_Float16*)nullptr, D);

  // ---- attention ----
  headsq_kernel<<<(N * H) / 8, blk, 0, stream>>>(bufQ, bufK, q2, k2);

  dim3 gQK(S / 64, S / 128, BH);
  gemm_nt_kernel<<<gQK, blk, 0, stream>>>(
      q16, D, (long)S * D, 64,
      k16, D, (long)S * D, 64,
      attn, S, (long)H * S * S, (long)S * S,
      64, H);

  attn_softmax_kernel<<<dim3(S, BH), blk, 0, stream>>>(attn, q2, k2, rsc);

  logmap0_head_kernel<<<(N * H) / 8, blk, 0, stream>>>(bufV, tv16);

  dim3 gAV(1, S / 128, BH);
  gemm_nn_kernel<<<gAV, blk, 0, stream>>>(
      attn, S, (long)H * S * S, (long)S * S,
      tv16, D, (long)S * D, 64,
      bufA, D, (long)S * D, 64,
      S, H);

  expmap0_head_kernel<<<(N * H) / 8, blk, 0, stream>>>(bufA, rsc, at16);

  // ---- output projection + residual + layernorm 1 ----
  rownorm_kernel<<<N, blk, 0, stream>>>(bufA, D, na);
  gemm_nt_kernel<<<gP, blk, 0, stream>>>(at16, D, 0, 0, wo16, D, 0, 0, buf0, D, 0, 0, D, 1);
  mobius_post_kernel<4><<<N, blk, 0, stream>>>(buf0, na, bo, bufO, (_Float16*)nullptr, D);
  resid_ln_kernel<<<N, blk, 0, stream>>>(x, bufO, g1, be1, hb1, bufQ, x116, n1);  // x1

  // ---- FFN ----
  dim3 gF1(F / 64, N / 128, 1);
  gemm_nt_kernel<<<gF1, blk, 0, stream>>>(x116, D, 0, 0, w116, D, 0, 0, buf0, F, 0, 0, D, 1);
  mobius_post_kernel<16><<<N, blk, 0, stream>>>(buf0, n1, b1, buf0, (_Float16*)nullptr, F);
  hyp_act_kernel<<<N, blk, 0, stream>>>(buf0, h16, nh);
  gemm_nt_kernel<<<gP, blk, 0, stream>>>(h16, F, 0, 0, w216, F, 0, 0, bufA, D, 0, 0, F, 1);
  mobius_post_kernel<4><<<N, blk, 0, stream>>>(bufA, nh, b2, bufA, (_Float16*)nullptr, D);

  // ---- residual + layernorm 2 -> final x ----
  resid_ln_kernel<<<N, blk, 0, stream>>>(bufQ, bufA, g2, be2, hb2, xout,
                                         (_Float16*)nullptr, (float*)nullptr);
}
